// PersistentMemory_7610682048659
// MI455X (gfx1250) — matrix operand order repaired and checked by reading, not compiled, not run on hardware
//
#include <hip/hip_runtime.h>
#include <hip/hip_bf16.h>
#include <math.h>

// ---------------------------------------------------------------------------
// Types
// ---------------------------------------------------------------------------
typedef __attribute__((ext_vector_type(16))) _Float16 v16h;
typedef __attribute__((ext_vector_type(8)))  _Float16 v8h;
typedef __attribute__((ext_vector_type(8)))  float    v8f;

#define DIM      1024
#define MEM_DIM  1024
#define BANKN    8192
#define NTOK     16384       // B*S = 4*4096
#define LN_EPS   1e-5f
#define SM_SCALE 0.03125f    // 1/sqrt(1024)

// ---------------------------------------------------------------------------
// CDNA5 async global->LDS copy (GLOBAL_LOAD_ASYNC_TO_LDS_B128, ASYNCcnt).
// VDST = per-lane LDS byte offset, VADDR = 64-bit global address.
// The low 32 bits of a generic pointer into LDS are the LDS byte offset
// (flat-aperture mapping, CDNA5 ISA).
// ---------------------------------------------------------------------------
__device__ __forceinline__ unsigned lds_off(const void* p) {
  return (unsigned)(unsigned long long)p;
}

__device__ __forceinline__ void async_load_b128(unsigned lds, const void* gaddr) {
  asm volatile("global_load_async_to_lds_b128 %0, %1, off"
               :: "v"(lds), "v"(gaddr) : "memory");
}

__device__ __forceinline__ void wait_async() {
#if __has_builtin(__builtin_amdgcn_s_wait_asynccnt)
  __builtin_amdgcn_s_wait_asynccnt(0);
#else
  asm volatile("s_wait_asynccnt 0x0" ::: "memory");
#endif
}

// ---------------------------------------------------------------------------
// WMMA helpers (CDNA5: V_WMMA_F32_16X16X32_F16, D = A(16x32) * B(32x16) + C)
// ---------------------------------------------------------------------------
__device__ __forceinline__ v8f wmma16(v16h a, v16h b, v8f c) {
  return __builtin_amdgcn_wmma_f32_16x16x32_f16(
      /*neg_a=*/false, a, /*neg_b=*/false, b,
      /*c_mod=*/(short)0, c, /*reuse_a=*/false, /*reuse_b=*/false);
}

// A fragment, 16(M) x 32(K) f16 from row-major source (ld in halves).
// ISA layout: lanes 0-15 -> row m=lane, k in [0..7] (v0-3) and [16..23] (v4-7);
//             lanes 16-31 -> row m=lane-16, k in [8..15] and [24..31].
__device__ __forceinline__ v16h load_a_frag(const _Float16* base, int ld) {
  int lane = threadIdx.x & 31;
  int row  = lane & 15;
  int k0   = (lane < 16) ? 0 : 8;
  const _Float16* p = base + row * ld + k0;
  union { v16h v; v8h h[2]; } u;
  u.h[0] = *(const v8h*)(p);
  u.h[1] = *(const v8h*)(p + 16);
  return u.v;
}

// Same as above but source is f32 (converted per element) — used for P tiles.
__device__ __forceinline__ v16h load_a_frag_f32(const float* base, int ld) {
  int lane = threadIdx.x & 31;
  int row  = lane & 15;
  int k0   = (lane < 16) ? 0 : 8;
  const float* p = base + row * ld + k0;
  union { v16h v; _Float16 e[16]; } u;
#pragma unroll
  for (int i = 0; i < 8; ++i) {
    u.e[i]     = (_Float16)p[i];
    u.e[8 + i] = (_Float16)p[16 + i];
  }
  return u.v;
}

// B fragment, 32(K) x 16(N): B[k][n] with lane n holding 16 k of
// column n. Source is a row-major matrix whose ROWS are n and COLUMNS are k
// (i.e. W[N][K] for C=A*W^T, or V^T[D][BANK] for P*V), ld in halves.
// lanes 0-15 -> n=lane,    k in [k0 .. k0+7] and [k0+16 .. k0+23]
// lanes 16-31 -> n=lane-16, k in [k0+8 .. k0+15] and [k0+24 .. k0+31]
__device__ __forceinline__ v16h load_b_frag(const _Float16* base, int ld) {
  int lane = threadIdx.x & 31;
  int n    = lane & 15;
  int ks   = (lane < 16) ? 0 : 8;
  const _Float16* p = base + n * ld + ks;
  union { v16h v; v8h h[2]; } u;
  u.h[0] = *(const v8h*)(p);
  u.h[1] = *(const v8h*)(p + 16);
  return u.v;
}

// ---------------------------------------------------------------------------
// f32 -> f16 conversion
// ---------------------------------------------------------------------------
__global__ void cvt_f32_to_f16(const float* __restrict__ src,
                               _Float16* __restrict__ dst, int n) {
  int i = blockIdx.x * blockDim.x + threadIdx.x;
  if (i < n) dst[i] = (_Float16)src[i];
}

// ---------------------------------------------------------------------------
// GEMM: C[M,N] = A[M,K] * B[N,K]^T + bias[N]   (all f16 in, f16 out, f32 acc)
// Block: 256 threads (8 waves, 2x4 wave grid), tile 128x128, k-step 32.
// Tile staging uses CDNA5 async global->LDS copies (no staging VGPRs).
// transposeOut != 0  ->  C stored transposed: C[n*M + m]  (used for V^T).
// ---------------------------------------------------------------------------
__global__ void gemm_abT_bias(const _Float16* __restrict__ A,
                              const _Float16* __restrict__ B,
                              const float* __restrict__ bias,
                              _Float16* __restrict__ C,
                              int M, int N, int K, int transposeOut) {
  __shared__ __attribute__((aligned(16))) _Float16 sA[128 * 32];
  __shared__ __attribute__((aligned(16))) _Float16 sB[128 * 32];

  const int tid  = threadIdx.x;
  const int wave = tid >> 5;
  const int lane = tid & 31;
  const int wm   = wave & 1;   // 0..1  (64-row slice)
  const int wn   = wave >> 1;  // 0..3  (32-col slice)
  const int m0   = blockIdx.x * 128;
  const int n0   = blockIdx.y * 128;

  v8f acc[4][2] = {};

  for (int k0 = 0; k0 < K; k0 += 32) {
    // Async stage: 128 rows x 32 halves each for A and B (4 x b128 per thread).
    {
      int r = tid >> 1;
      int c = (tid & 1) * 16;
      async_load_b128(lds_off(&sA[r * 32 + c]),
                      &A[(size_t)(m0 + r) * K + k0 + c]);
      async_load_b128(lds_off(&sA[r * 32 + c + 8]),
                      &A[(size_t)(m0 + r) * K + k0 + c + 8]);
      async_load_b128(lds_off(&sB[r * 32 + c]),
                      &B[(size_t)(n0 + r) * K + k0 + c]);
      async_load_b128(lds_off(&sB[r * 32 + c + 8]),
                      &B[(size_t)(n0 + r) * K + k0 + c + 8]);
    }
    wait_async();
    __syncthreads();

    v16h bf0 = load_b_frag(&sB[(wn * 32 + 0)  * 32], 32);
    v16h bf1 = load_b_frag(&sB[(wn * 32 + 16) * 32], 32);
#pragma unroll
    for (int i = 0; i < 4; ++i) {
      v16h af = load_a_frag(&sA[(wm * 64 + i * 16) * 32], 32);
      acc[i][0] = wmma16(af, bf0, acc[i][0]);
      acc[i][1] = wmma16(af, bf1, acc[i][1]);
    }
    __syncthreads();
  }

  // Epilogue: C layout — lane holds column (lane&15), VGPR r holds row r / r+8.
  const int col = lane & 15;
  const int rb  = (lane < 16) ? 0 : 8;
#pragma unroll
  for (int i = 0; i < 4; ++i) {
#pragma unroll
    for (int j = 0; j < 2; ++j) {
      int gm = m0 + wm * 64 + i * 16 + rb;
      int gn = n0 + wn * 32 + j * 16 + col;
      float bv = bias[gn];
#pragma unroll
      for (int r = 0; r < 8; ++r) {
        float v = acc[i][j][r] + bv;
        if (!transposeOut)
          C[(size_t)(gm + r) * N + gn] = (_Float16)v;
        else
          C[(size_t)gn * M + (gm + r)] = (_Float16)v;
      }
    }
  }
}

// ---------------------------------------------------------------------------
// Fused streaming attention over the knowledge bank (flash-style).
// Block: 32 query rows, 256 threads (8 waves). Q tile resident in LDS
// (loaded via async global->LDS). Per bank-tile of 128: S = Q*K^T (each wave
// owns 16 score columns), online softmax stats in LDS, then O += P*V^T per
// wave d-slice of 128.
// Dynamic LDS: sQ 64KB | sS 16KB | red 1KB | m/l/alpha 384B.
// ---------------------------------------------------------------------------
__global__ void attention_bank(const _Float16* __restrict__ Qh,   // [NTOK, 1024]
                               const _Float16* __restrict__ Kh,   // [8192, 1024]
                               const _Float16* __restrict__ Vth,  // [1024, 8192] (V^T)
                               _Float16* __restrict__ Rh) {       // [NTOK, 1024]
  extern __shared__ __attribute__((aligned(16))) char smem[];
  _Float16* sQ    = (_Float16*)smem;                        // 32*1024 halves
  float*    sS    = (float*)(smem + 65536);                 // 32*128 f32
  float*    red   = (float*)(smem + 65536 + 16384);         // 32*8
  float*    rowM  = (float*)(smem + 65536 + 16384 + 1024);  // 32
  float*    rowL  = rowM + 32;                              // 32
  float*    alpha = rowL + 32;                              // 32

  const int tid  = threadIdx.x;
  const int wave = tid >> 5;
  const int lane = tid & 31;
  const int q0   = blockIdx.x * 32;
  const int d0   = wave * 128;          // wave-owned output d-slice
  const int col  = lane & 15;
  const int rb   = (lane < 16) ? 0 : 8;

  // Stage the 32x1024 Q tile (rows contiguous in memory) with async copies.
  {
    const _Float16* g = &Qh[(size_t)q0 * 1024];
#pragma unroll
    for (int i = 0; i < 16; ++i) {  // 4096 b128 chunks / 256 threads
      int chunk = tid + i * 256;    // 16-byte chunk index
      async_load_b128(lds_off(&sQ[chunk * 8]), g + chunk * 8);
    }
  }
  if (tid < 32) { rowM[tid] = -3.0e38f; rowL[tid] = 0.0f; }
  wait_async();
  __syncthreads();

  v8f o[2][8] = {};   // O accumulator: 32 rows x 128 cols per wave

  for (int nk0 = 0; nk0 < BANKN; nk0 += 128) {
    // Prefetch next K tile (global_prefetch_b8).
    if (nk0 + 128 < BANKN)
      __builtin_prefetch(&Kh[(size_t)(nk0 + 128 + wave * 16 + col) * 1024], 0, 0);

    // ---- S = Q * K^T : wave computes 32 rows x 16 cols (cols nk0+wave*16) --
    v8f s[2] = {};
    for (int k0 = 0; k0 < MEM_DIM; k0 += 32) {
      v16h b  = load_b_frag(&Kh[(size_t)(nk0 + wave * 16) * 1024 + k0], 1024);
      v16h a0 = load_a_frag(&sQ[0 * 1024 + k0], 1024);
      v16h a1 = load_a_frag(&sQ[16 * 1024 + k0], 1024);
      s[0] = wmma16(a0, b, s[0]);
      s[1] = wmma16(a1, b, s[1]);
    }
#pragma unroll
    for (int qf = 0; qf < 2; ++qf)
#pragma unroll
      for (int r = 0; r < 8; ++r)
        sS[(qf * 16 + rb + r) * 128 + wave * 16 + col] = s[qf][r] * SM_SCALE;
    __syncthreads();

    // ---- online softmax stats: pass 1 — per-segment row max ----------------
    {
      int row = tid & 31, seg = tid >> 5;
      float pm = -3.0e38f;
#pragma unroll
      for (int c = 0; c < 16; ++c)
        pm = fmaxf(pm, sS[row * 128 + seg * 16 + c]);
      red[row * 8 + seg] = pm;
    }
    __syncthreads();
    if (tid < 32) {
      float tm = red[tid * 8];
#pragma unroll
      for (int g = 1; g < 8; ++g) tm = fmaxf(tm, red[tid * 8 + g]);
      float mOld = rowM[tid];
      float mNew = fmaxf(mOld, tm);
      alpha[tid] = __expf(mOld - mNew);
      rowM[tid]  = mNew;
    }
    __syncthreads();

    // ---- pass 2 — exponentiate in place, per-segment partial sums ----------
    {
      int row = tid & 31, seg = tid >> 5;
      float mr = rowM[row];
      float ps = 0.0f;
#pragma unroll
      for (int c = 0; c < 16; ++c) {
        float v = __expf(sS[row * 128 + seg * 16 + c] - mr);
        sS[row * 128 + seg * 16 + c] = v;
        ps += v;
      }
      red[row * 8 + seg] = ps;
    }
    __syncthreads();
    if (tid < 32) {
      float sum = 0.0f;
#pragma unroll
      for (int g = 0; g < 8; ++g) sum += red[tid * 8 + g];
      rowL[tid] = rowL[tid] * alpha[tid] + sum;
    }
    __syncthreads();

    // ---- O = O*alpha + P*V : per wave d-slice of 128 -----------------------
#pragma unroll
    for (int qf = 0; qf < 2; ++qf)
#pragma unroll
      for (int r = 0; r < 8; ++r) {
        float al = alpha[qf * 16 + rb + r];
#pragma unroll
        for (int df = 0; df < 8; ++df) o[qf][df][r] *= al;
      }
#pragma unroll
    for (int kk = 0; kk < 4; ++kk) {
      v16h a0 = load_a_frag_f32(&sS[0 * 128 + kk * 32], 128);
      v16h a1 = load_a_frag_f32(&sS[16 * 128 + kk * 32], 128);
#pragma unroll
      for (int df = 0; df < 8; ++df) {
        v16h b = load_b_frag(&Vth[(size_t)(d0 + df * 16) * BANKN + nk0 + kk * 32], BANKN);
        o[0][df] = wmma16(a0, b, o[0][df]);
        o[1][df] = wmma16(a1, b, o[1][df]);
      }
    }
    __syncthreads();   // sS reused next iteration
  }

  // ---- finalize: retrieved = O / l -----------------------------------------
#pragma unroll
  for (int qf = 0; qf < 2; ++qf)
#pragma unroll
    for (int r = 0; r < 8; ++r) {
      int row = qf * 16 + rb + r;
      float rl = 1.0f / rowL[row];
#pragma unroll
      for (int df = 0; df < 8; ++df)
        Rh[(size_t)(q0 + row) * 1024 + d0 + df * 16 + col] =
            (_Float16)(o[qf][df][r] * rl);
    }
}

// ---------------------------------------------------------------------------
// Fused output projection + bias + residual + LayerNorm.
// out[t,n] = LN( [x||retrieved][t,:] . wf[n,:] + bf[n] + x[t,n] )
// Block: 32 tokens, 256 threads; each wave owns 128 of the 1024 out columns.
// ---------------------------------------------------------------------------
__global__ void out_proj_ln(const _Float16* __restrict__ xh,   // [NTOK,1024]
                            const _Float16* __restrict__ Rh,   // [NTOK,1024]
                            const _Float16* __restrict__ wfh,  // [1024,2048]
                            const float* __restrict__ bf,
                            const float* __restrict__ x,       // f32 residual
                            const float* __restrict__ gamma,
                            const float* __restrict__ beta,
                            float* __restrict__ out) {
  __shared__ __attribute__((aligned(16))) _Float16 sA[32 * 32];
  __shared__ float ssum[32], ssq[32];

  const int tid  = threadIdx.x;
  const int wave = tid >> 5;
  const int lane = tid & 31;
  const int q0   = blockIdx.x * 32;
  const int d0   = wave * 128;
  const int col  = lane & 15;
  const int rb   = (lane < 16) ? 0 : 8;

  if (tid < 32) { ssum[tid] = 0.0f; ssq[tid] = 0.0f; }

  v8f acc[2][8] = {};

  for (int k0 = 0; k0 < DIM + MEM_DIM; k0 += 32) {
    if (tid < 64) {  // stage 32x32 combined-A slab via async copies
      int r = tid >> 1, c = (tid & 1) * 16;
      const _Float16* src = (k0 < DIM)
          ? &xh[(size_t)(q0 + r) * 1024 + k0 + c]
          : &Rh[(size_t)(q0 + r) * 1024 + (k0 - DIM) + c];
      async_load_b128(lds_off(&sA[r * 32 + c]), src);
      async_load_b128(lds_off(&sA[r * 32 + c + 8]), src + 8);
    }
    wait_async();
    __syncthreads();

    v16h a0 = load_a_frag(&sA[0],       32);
    v16h a1 = load_a_frag(&sA[16 * 32], 32);
#pragma unroll
    for (int df = 0; df < 8; ++df) {
      v16h b = load_b_frag(&wfh[(size_t)(d0 + df * 16) * 2048 + k0], 2048);
      acc[0][df] = wmma16(a0, b, acc[0][df]);
      acc[1][df] = wmma16(a1, b, acc[1][df]);
    }
    __syncthreads();
  }

  // h = acc + bf + x ; accumulate LN stats with LDS float atomics (ds_add_f32)
#pragma unroll
  for (int qf = 0; qf < 2; ++qf) {
#pragma unroll
    for (int r = 0; r < 8; ++r) {
      int gm = q0 + qf * 16 + rb + r;
      float psum = 0.0f, psq = 0.0f;
#pragma unroll
      for (int df = 0; df < 8; ++df) {
        int gn = d0 + df * 16 + col;
        float v = acc[qf][df][r] + bf[gn] + x[(size_t)gm * 1024 + gn];
        acc[qf][df][r] = v;
        psum += v;
        psq  += v * v;
      }
      atomicAdd(&ssum[qf * 16 + rb + r], psum);
      atomicAdd(&ssq[qf * 16 + rb + r], psq);
    }
  }
  __syncthreads();

#pragma unroll
  for (int qf = 0; qf < 2; ++qf) {
#pragma unroll
    for (int r = 0; r < 8; ++r) {
      int row = qf * 16 + rb + r;
      float mu  = ssum[row] * (1.0f / 1024.0f);
      float var = ssq[row] * (1.0f / 1024.0f) - mu * mu;
      float rs  = rsqrtf(var + LN_EPS);
#pragma unroll
      for (int df = 0; df < 8; ++df) {
        int gn = d0 + df * 16 + col;
        out[(size_t)(q0 + row) * 1024 + gn] =
            (acc[qf][df][r] - mu) * rs * gamma[gn] + beta[gn];
      }
    }
  }
}

// ---------------------------------------------------------------------------
// Host-side launch
// ---------------------------------------------------------------------------
extern "C" void kernel_launch(void* const* d_in, const int* in_sizes, int n_in,
                              void* d_out, int out_size, void* d_ws, size_t ws_size,
                              hipStream_t stream) {
  const float* x    = (const float*)d_in[0];   // [4,4096,1024]
  const float* bank = (const float*)d_in[1];   // [8192,1024]
  const float* wq   = (const float*)d_in[2];   // [1024,1024]
  const float* bq   = (const float*)d_in[3];
  const float* wk   = (const float*)d_in[4];
  const float* bk   = (const float*)d_in[5];
  const float* wv   = (const float*)d_in[6];
  const float* bv   = (const float*)d_in[7];
  const float* wf   = (const float*)d_in[8];   // [1024,2048]
  const float* bf   = (const float*)d_in[9];
  const float* gamma= (const float*)d_in[10];
  const float* beta = (const float*)d_in[11];
  float* out = (float*)d_out;

  // Workspace layout (bytes)
  char* ws = (char*)d_ws;
  _Float16* xh    = (_Float16*)(ws);                  // 33,554,432
  _Float16* bankh = (_Float16*)(ws + 33554432ULL);    // 16,777,216
  _Float16* wqh   = (_Float16*)(ws + 50331648ULL);    //  2,097,152
  _Float16* wkh   = (_Float16*)(ws + 52428800ULL);    //  2,097,152
  _Float16* wvh   = (_Float16*)(ws + 54525952ULL);    //  2,097,152
  _Float16* wfh   = (_Float16*)(ws + 56623104ULL);    //  4,194,304
  _Float16* Qh    = (_Float16*)(ws + 60817408ULL);    // 33,554,432
  _Float16* Kh    = (_Float16*)(ws + 94371840ULL);    // 16,777,216
  _Float16* Vth   = (_Float16*)(ws + 111149056ULL);   // 16,777,216 (V^T)
  _Float16* Rh    = (_Float16*)(ws + 127926272ULL);   // 33,554,432

  // 1) f32 -> f16 conversions
  cvt_f32_to_f16<<<NTOK * DIM / 256, 256, 0, stream>>>(x, xh, NTOK * DIM);
  cvt_f32_to_f16<<<BANKN * MEM_DIM / 256, 256, 0, stream>>>(bank, bankh, BANKN * MEM_DIM);
  cvt_f32_to_f16<<<1024 * 1024 / 256, 256, 0, stream>>>(wq, wqh, 1024 * 1024);
  cvt_f32_to_f16<<<1024 * 1024 / 256, 256, 0, stream>>>(wk, wkh, 1024 * 1024);
  cvt_f32_to_f16<<<1024 * 1024 / 256, 256, 0, stream>>>(wv, wvh, 1024 * 1024);
  cvt_f32_to_f16<<<1024 * 2048 / 256, 256, 0, stream>>>(wf, wfh, 1024 * 2048);

  // 2) Projections (WMMA GEMMs)
  gemm_abT_bias<<<dim3(NTOK / 128, MEM_DIM / 128), 256, 0, stream>>>(
      xh, wqh, bq, Qh, NTOK, MEM_DIM, DIM, 0);
  gemm_abT_bias<<<dim3(BANKN / 128, MEM_DIM / 128), 256, 0, stream>>>(
      bankh, wkh, bk, Kh, BANKN, MEM_DIM, MEM_DIM, 0);
  gemm_abT_bias<<<dim3(BANKN / 128, DIM / 128), 256, 0, stream>>>(
      bankh, wvh, bv, Vth, BANKN, DIM, MEM_DIM, 1 /* store V^T */);

  // 3) Fused attention over the bank (dynamic LDS: 83,328 bytes)
  size_t attn_lds = 65536 + 16384 + 1024 + 3 * 32 * 4;
  attention_bank<<<NTOK / 32, 256, attn_lds, stream>>>(Qh, Kh, Vth, Rh);

  // 4) Output projection + residual + LayerNorm
  out_proj_ln<<<NTOK / 32, 256, 0, stream>>>(xh, Rh, wfh, bf, x, gamma, beta, out);
}
